// SelfAttention_41394894799048
// MI455X (gfx1250) — matrix operand order repaired and checked by reading, not compiled, not run on hardware
//
#include <hip/hip_runtime.h>

#define E_DIM 1024
#define S_LEN 2048
#define NB 4
#define NH 16
#define DH 64
#define BS (NB * S_LEN) // 8192 total rows

typedef __attribute__((ext_vector_type(16))) __bf16 v16bf;
typedef __attribute__((ext_vector_type(8)))  __bf16 v8bf;
typedef __attribute__((ext_vector_type(4)))  __bf16 v4bf;
typedef __attribute__((ext_vector_type(8)))  float  v8f;

static __device__ __forceinline__ v16bf cat8(v8bf a, v8bf b) {
  return __builtin_shufflevector(a, b, 0,1,2,3,4,5,6,7,8,9,10,11,12,13,14,15);
}

// A operand (16x32 bf16), row-major source. Lane holds row M = lane&15.
// elem e: K = k0 + hi*16 + e  (contiguous).
static __device__ __forceinline__ v16bf load_a32(const __bf16* row, int k0, int hi) {
  return *(const v16bf*)(row + k0 + hi * 16);
}

// B operand (32x16 bf16) given B^T stored row-major (rowBT = &BT[n][0]).
// Lane holds col N = lane&15; elem e: K = k0 + hi*16 + e  (contiguous).
static __device__ __forceinline__ v16bf load_b32(const __bf16* rowBT, int k0, int hi) {
  return *(const v16bf*)(rowBT + k0 + hi * 16);
}

static __device__ __forceinline__ v8f wmma_bf16(v16bf a, v16bf b, v8f c) {
  return __builtin_amdgcn_wmma_f32_16x16x32_bf16(false, a, false, b, (short)0, c,
                                                 false, false);
}

// ---------------------------------------------------------------- cvt kernel
__global__ __launch_bounds__(256) void cvt_f32_bf16(const float* __restrict__ in,
                                                    __bf16* __restrict__ out, int n4) {
  int i = blockIdx.x * 256 + threadIdx.x;
  if (i < n4) {
    float4 v = ((const float4*)in)[i];
    v4bf o;
    o[0] = (__bf16)v.x; o[1] = (__bf16)v.y; o[2] = (__bf16)v.z; o[3] = (__bf16)v.w;
    ((v4bf*)out)[i] = o;
  }
}

// ------------------------------------------------------------- fused QKV GEMM
// Q = x@wq^T + bq (pre-scaled by 1/8), K = x@wk^T + bk, V = x@wv^T + bv.
// Q,K stored [B,H,S,Dh]; V stored transposed [B,H,Dh,S].
__global__ __launch_bounds__(256) void qkv_proj(
    const __bf16* __restrict__ xb,
    const __bf16* __restrict__ wqb, const __bf16* __restrict__ wkb,
    const __bf16* __restrict__ wvb,
    const float* __restrict__ bq, const float* __restrict__ bk,
    const float* __restrict__ bv,
    __bf16* __restrict__ qo, __bf16* __restrict__ ko, __bf16* __restrict__ vto)
{
  const int wave = blockIdx.x * 8 + (threadIdx.x >> 5);
  const int lane = threadIdx.x & 31;
  const int hi = lane >> 4, col = lane & 15;
  const int m0 = (wave >> 4) * 16;   // 512 row tiles
  const int n0 = (wave & 15) * 64;   // 16 col tiles of 64
  const __bf16* arow = xb + (size_t)(m0 + col) * E_DIM;

  v8f acc[3][4];
  #pragma unroll
  for (int t = 0; t < 3; ++t)
    #pragma unroll
    for (int j = 0; j < 4; ++j) acc[t][j] = {};

  for (int k0 = 0; k0 < E_DIM; k0 += 32) {
    v16bf a = load_a32(arow, k0, hi);
    #pragma unroll
    for (int j = 0; j < 4; ++j) {
      const size_t n = (size_t)(n0 + j * 16 + col);
      acc[0][j] = wmma_bf16(a, load_b32(wqb + n * E_DIM, k0, hi), acc[0][j]);
      acc[1][j] = wmma_bf16(a, load_b32(wkb + n * E_DIM, k0, hi), acc[1][j]);
      acc[2][j] = wmma_bf16(a, load_b32(wvb + n * E_DIM, k0, hi), acc[2][j]);
    }
  }

  #pragma unroll
  for (int j = 0; j < 4; ++j) {
    const int e = n0 + j * 16 + col;
    const int h = e >> 6, d = e & 63;
    const float zq = bq[e], zk = bk[e], zv = bv[e];
    #pragma unroll
    for (int r = 0; r < 8; ++r) {
      const int m = m0 + r + 8 * hi;
      const int b = m >> 11, s = m & (S_LEN - 1);
      const size_t bh = (size_t)(b * NH + h);
      qo[(bh * S_LEN + s) * DH + d] = (__bf16)((acc[0][j][r] + zq) * 0.125f);
      ko[(bh * S_LEN + s) * DH + d] = (__bf16)(acc[1][j][r] + zk);
      vto[(bh * DH + d) * S_LEN + s] = (__bf16)(acc[2][j][r] + zv);
    }
  }
}

// ------------------------------------------------- flash attention (causal)
__global__ __launch_bounds__(256) void attn_fwd(
    const __bf16* __restrict__ qb, const __bf16* __restrict__ kb,
    const __bf16* __restrict__ vtb, __bf16* __restrict__ ctxb)
{
  __shared__ __bf16 plds[8 * 16 * 32]; // 1 KiB per wave
  const int wave = threadIdx.x >> 5, lane = threadIdx.x & 31;
  const int hi = lane >> 4, col = lane & 15;
  const int bh = blockIdx.x >> 4;                 // (b*H + h)
  const int m0 = (blockIdx.x & 15) * 128 + wave * 16;
  const __bf16* Q  = qb  + (size_t)bh * S_LEN * DH;
  const __bf16* K  = kb  + (size_t)bh * S_LEN * DH;
  const __bf16* Vt = vtb + (size_t)bh * DH * S_LEN;
  __bf16* lds = plds + wave * (16 * 32);

  const __bf16* qrow = Q + (size_t)(m0 + col) * DH;
  const v16bf aq0 = load_a32(qrow, 0, hi);
  const v16bf aq1 = load_a32(qrow, 32, hi);

  v8f o[4];
  #pragma unroll
  for (int j = 0; j < 4; ++j) o[j] = {};
  float mrow[8], lrow[8];
  #pragma unroll
  for (int r = 0; r < 8; ++r) { mrow[r] = -__builtin_inff(); lrow[r] = 0.0f; }

  for (int t0 = 0; t0 <= m0 + 15; t0 += 32) {
    const __bf16* kr0 = K + (size_t)(t0 + col) * DH;
    const __bf16* kr1 = K + (size_t)(t0 + 16 + col) * DH;
    v8f s0 = {}, s1 = {};
    s0 = wmma_bf16(aq0, load_b32(kr0, 0, hi), s0);
    s0 = wmma_bf16(aq1, load_b32(kr0, 32, hi), s0);
    s1 = wmma_bf16(aq0, load_b32(kr1, 0, hi), s1);
    s1 = wmma_bf16(aq1, load_b32(kr1, 32, hi), s1);

    const bool partial = (t0 + 31 > m0);
    #pragma unroll
    for (int r = 0; r < 8; ++r) {
      const int mr = m0 + r + 8 * hi;
      float a0 = s0[r], a1 = s1[r];
      if (partial) {
        if (t0 + col > mr)      a0 = -__builtin_inff();
        if (t0 + 16 + col > mr) a1 = -__builtin_inff();
      }
      float tmax = fmaxf(a0, a1);
      tmax = fmaxf(tmax, __shfl_xor(tmax, 1));
      tmax = fmaxf(tmax, __shfl_xor(tmax, 2));
      tmax = fmaxf(tmax, __shfl_xor(tmax, 4));
      tmax = fmaxf(tmax, __shfl_xor(tmax, 8));
      const float mnew  = fmaxf(mrow[r], tmax);
      const float alpha = __expf(mrow[r] - mnew);
      const float p0 = __expf(a0 - mnew);
      const float p1 = __expf(a1 - mnew);
      float ps = p0 + p1;
      ps += __shfl_xor(ps, 1); ps += __shfl_xor(ps, 2);
      ps += __shfl_xor(ps, 4); ps += __shfl_xor(ps, 8);
      lrow[r] = lrow[r] * alpha + ps;
      mrow[r] = mnew;
      #pragma unroll
      for (int j = 0; j < 4; ++j) o[j][r] *= alpha;
      // C-layout -> row-major 16x32 bf16 tile in LDS
      lds[(r + 8 * hi) * 32 + col]      = (__bf16)p0;
      lds[(r + 8 * hi) * 32 + 16 + col] = (__bf16)p1;
    }
    asm volatile("s_wait_dscnt 0" ::: "memory"); // wave-level LDS fence
    // read back in WMMA A layout (row M = col, one contiguous 16-elem chunk)
    const v16bf ap = *(const v16bf*)(lds + col * 32 + hi * 16);
    #pragma unroll
    for (int j = 0; j < 4; ++j) {
      const v16bf bv = *(const v16bf*)(Vt + (size_t)(j * 16 + col) * S_LEN + t0 + hi * 16);
      o[j] = wmma_bf16(ap, bv, o[j]);
    }
  }

  const int b = bh >> 4, h = bh & 15;
  #pragma unroll
  for (int j = 0; j < 4; ++j) {
    #pragma unroll
    for (int r = 0; r < 8; ++r) {
      const int mr = m0 + r + 8 * hi;
      const float inv = 1.0f / lrow[r];
      ctxb[((size_t)(b * S_LEN + mr)) * E_DIM + h * DH + j * 16 + col] =
          (__bf16)(o[j][r] * inv);
    }
  }
}

// --------------------------------------------------------- output projection
__global__ __launch_bounds__(256) void out_proj(
    const __bf16* __restrict__ ctxb, const __bf16* __restrict__ wob,
    const float* __restrict__ bo, float* __restrict__ out)
{
  const int wave = blockIdx.x * 8 + (threadIdx.x >> 5);
  const int lane = threadIdx.x & 31;
  const int hi = lane >> 4, col = lane & 15;
  const int m0 = (wave >> 4) * 16;
  const int n0 = (wave & 15) * 64;
  const __bf16* arow = ctxb + (size_t)(m0 + col) * E_DIM;

  v8f acc[4];
  #pragma unroll
  for (int j = 0; j < 4; ++j) acc[j] = {};
  for (int k0 = 0; k0 < E_DIM; k0 += 32) {
    v16bf a = load_a32(arow, k0, hi);
    #pragma unroll
    for (int j = 0; j < 4; ++j) {
      const size_t n = (size_t)(n0 + j * 16 + col);
      acc[j] = wmma_bf16(a, load_b32(wob + n * E_DIM, k0, hi), acc[j]);
    }
  }
  #pragma unroll
  for (int j = 0; j < 4; ++j) {
    const int e = n0 + j * 16 + col;
    const float z = bo[e];
    #pragma unroll
    for (int r = 0; r < 8; ++r) {
      const int m = m0 + r + 8 * hi;
      out[(size_t)m * E_DIM + e] = acc[j][r] + z;
    }
  }
}

// ----------------------------------------------------------------- launcher
extern "C" void kernel_launch(void* const* d_in, const int* in_sizes, int n_in,
                              void* d_out, int out_size, void* d_ws, size_t ws_size,
                              hipStream_t stream)
{
  (void)in_sizes; (void)n_in; (void)out_size; (void)ws_size;
  const float* x  = (const float*)d_in[0];
  const float* wq = (const float*)d_in[1];
  const float* bq = (const float*)d_in[2];
  const float* wk = (const float*)d_in[3];
  const float* bk = (const float*)d_in[4];
  const float* wv = (const float*)d_in[5];
  const float* bv = (const float*)d_in[6];
  const float* wo = (const float*)d_in[7];
  const float* bo = (const float*)d_in[8];
  float* out = (float*)d_out;

  char* ws = (char*)d_ws;
  const size_t XB = (size_t)BS * E_DIM * 2;   // 16 MiB per [BS,E] bf16 buffer
  const size_t WB = (size_t)E_DIM * E_DIM * 2; // 2 MiB per weight
  __bf16* xb   = (__bf16*)(ws);
  __bf16* wqb  = (__bf16*)(ws + XB);
  __bf16* wkb  = (__bf16*)(ws + XB + 1 * WB);
  __bf16* wvb  = (__bf16*)(ws + XB + 2 * WB);
  __bf16* wob  = (__bf16*)(ws + XB + 3 * WB);
  __bf16* qb   = (__bf16*)(ws + 1 * XB + 4 * WB);
  __bf16* kb   = (__bf16*)(ws + 2 * XB + 4 * WB);
  __bf16* vtb  = (__bf16*)(ws + 3 * XB + 4 * WB);
  __bf16* ctxb = (__bf16*)(ws + 4 * XB + 4 * WB);

  const int nx4 = BS * E_DIM / 4;      // 2097152
  const int nw4 = E_DIM * E_DIM / 4;   // 262144
  cvt_f32_bf16<<<nx4 / 256, 256, 0, stream>>>(x,  xb,  nx4);
  cvt_f32_bf16<<<nw4 / 256, 256, 0, stream>>>(wq, wqb, nw4);
  cvt_f32_bf16<<<nw4 / 256, 256, 0, stream>>>(wk, wkb, nw4);
  cvt_f32_bf16<<<nw4 / 256, 256, 0, stream>>>(wv, wvb, nw4);
  cvt_f32_bf16<<<nw4 / 256, 256, 0, stream>>>(wo, wob, nw4);

  qkv_proj<<<1024, 256, 0, stream>>>(xb, wqb, wkb, wvb, bq, bk, bv, qb, kb, vtb);
  attn_fwd<<<1024, 256, 0, stream>>>(qb, kb, vtb, ctxb);
  out_proj<<<1024, 256, 0, stream>>>(ctxb, wob, bo, out);
}
